// AttentionDecoder_41283225649297
// MI455X (gfx1250) — compile-verified, matrix operand order checked by reading, not run on hardware
//
#include <hip/hip_runtime.h>
#include <hip/hip_bf16.h>
#include <math.h>

// ---------------------------------------------------------------------------
// AttentionDecoder for MI455X (gfx1250, wave32, WMMA)
//   B=32, T_DEC=256, T_ENC=256, E=128, H=256, V=32000
// Strategy:
//   - all GEMMs via v_wmma_f32_16x16x32_bf16 (bf16 in, f32 accumulate)
//   - GRU: 32 sync-free per-batch workgroups (serial in t), f32 state in LDS,
//     bf16 weights streamed from L2
//   - logits (268 GFLOP, 1.05GB out): 128x128 tiles, K=512 fully LDS-resident,
//     staged with the Tensor Data Mover (tensor_load_to_lds + s_wait_tensorcnt),
//     TDM pad_enable gives a 520-element padded LDS row stride (bank-conflict free)
// ---------------------------------------------------------------------------

typedef __attribute__((ext_vector_type(16))) __bf16 v16bf;
typedef __attribute__((ext_vector_type(8)))  float  v8f;
typedef __attribute__((ext_vector_type(4)))  unsigned int u32x4;
typedef __attribute__((ext_vector_type(8)))  int    i32x8;
typedef __attribute__((ext_vector_type(4)))  int    i32x4;

#define NB 32
#define TD 256
#define TE 256
#define EE 128
#define HH 256
#define VV 32000

__device__ __forceinline__ unsigned short f2bf_bits(float f) {
  return __builtin_bit_cast(unsigned short, (__bf16)f);
}
__device__ __forceinline__ float bfbits2f(unsigned short u) {
  return __builtin_bit_cast(float, ((unsigned)u) << 16);
}

__device__ __forceinline__ v8f wmma_bf16(v16bf a, v16bf b, v8f c) {
  // D = A(16x32) * B(32x16) + C, f32 accumulate
  return __builtin_amdgcn_wmma_f32_16x16x32_bf16(false, a, false, b, (short)0, c,
                                                 false, false);
}

// A/B fragment from a row-major bf16 matrix: lane holds row (row0 + lane&15),
// two contiguous 16B chunks per the CDNA5 16-bit A layout.
__device__ __forceinline__ v16bf load_frag(const unsigned short* base, int stride,
                                           int row0, int k0, int lane) {
  int r  = row0 + (lane & 15);
  int ko = (lane & 16) ? 8 : 0;
  const unsigned short* p = base + (long long)r * stride + k0 + ko;
  union { uint4 q[2]; v16bf v; } u;
  u.q[0] = *(const uint4*)p;
  u.q[1] = *(const uint4*)(p + 16);
  return u.v;
}

// fragment where each lane supplies its own row pointer (gathered rows)
__device__ __forceinline__ v16bf load_frag_rowptr(const unsigned short* rowbase,
                                                  int k0, int lane) {
  int ko = (lane & 16) ? 8 : 0;
  const unsigned short* p = rowbase + k0 + ko;
  union { uint4 q[2]; v16bf v; } u;
  u.q[0] = *(const uint4*)p;
  u.q[1] = *(const uint4*)(p + 16);
  return u.v;
}

// ---------------------------------------------------------------------------
// Tensor Data Mover: 2D tile (rows of 1024B, data_size=8B) global -> LDS,
// LDS padding 16B per 1024B row (pad_interval code 7 = 256 dwords,
// pad_amount code 3 = 4 dwords).  6-arg builtin variant (this toolchain).
// ---------------------------------------------------------------------------
__device__ __forceinline__ void tdm_load_tile_1024B_rows(unsigned lds_off,
                                                         const void* gptr,
                                                         unsigned nrows) {
  unsigned long long ga = (unsigned long long)(uintptr_t)gptr;
  u32x4 g0;
  g0[0] = 1u;                                     // count=1, user descriptor
  g0[1] = lds_off;                                // lds_addr
  g0[2] = (unsigned)(ga & 0xffffffffu);           // global_addr[31:0]
  g0[3] = (unsigned)((ga >> 32) & 0x1ffffffu) | (2u << 30);  // addr[56:32] | type=2

  const unsigned dim0 = 128;                      // 1024B row in 8B units
  i32x8 g1;
  g1[0] = (int)((3u << 16) | (1u << 20) | (7u << 22) | (3u << 25));
  //            data_size=8B  pad_en      pad_int=256dw pad_amt=4dw
  g1[1] = (int)((dim0 & 0xffffu) << 16);          // tensor_dim0 lo16
  g1[2] = (int)(((dim0 >> 16) & 0xffffu) | ((nrows & 0xffffu) << 16)); // dim0 hi | dim1 lo
  g1[3] = (int)(((nrows >> 16) & 0xffffu) | ((dim0 & 0xffffu) << 16)); // dim1 hi | tile_dim0
  g1[4] = (int)(nrows & 0xffffu);                 // tile_dim1 (tile_dim2=0)
  g1[5] = (int)dim0;                              // tensor_dim0_stride lo32 (=row)
  g1[6] = 0;                                      // stride hi16 | dim1_stride lo16
  g1[7] = 0;
  i32x4 z4 = {0, 0, 0, 0};
  i32x8 z8 = {0, 0, 0, 0, 0, 0, 0, 0};
  __builtin_amdgcn_tensor_load_to_lds(g0, g1, z4, z4, z8, 0);
}

// ---------------------------------------------------------------------------
// f32 -> bf16 convert
// ---------------------------------------------------------------------------
__global__ void __launch_bounds__(256) convert_bf16(const float* __restrict__ src,
                                                    unsigned short* __restrict__ dst,
                                                    int n) {
  int i = blockIdx.x * 256 + threadIdx.x;
  if (i < n) dst[i] = f2bf_bits(src[i]);
}

// hsT[b][h][e] = hs[b][e][h]  (bf16)
__global__ void __launch_bounds__(256) transpose_hs_kernel(const float* __restrict__ hs,
                                                           unsigned short* __restrict__ hsT) {
  int i = blockIdx.x * 256 + threadIdx.x;
  if (i < NB * TE * HH) {
    int b = i >> 16, e = (i >> 8) & 255, h = i & 255;
    hsT[((long long)b << 16) | (h << 8) | e] = f2bf_bits(hs[i]);
  }
}

// ---------------------------------------------------------------------------
// gx[b,t,g] = emb[seq[b,t]] @ w_ih[g,:]^T + b_ih[g]     (8192 x 768, K=128)
// one wave per 16x16 tile, bf16 WMMA
// ---------------------------------------------------------------------------
__global__ void __launch_bounds__(256) gx_kernel(const int* __restrict__ seq,
                                                 const unsigned short* __restrict__ embb,
                                                 const unsigned short* __restrict__ wihb,
                                                 const float* __restrict__ bih,
                                                 float* __restrict__ gx) {
  int lane = threadIdx.x & 31;
  int wave = threadIdx.x >> 5;
  int nt = blockIdx.x;                 // 0..47  (768/16)
  int mt = blockIdx.y * 8 + wave;      // 0..511 (8192/16)

  int tok = seq[mt * 16 + (lane & 15)];
  const unsigned short* arow = embb + (long long)tok * EE;

  v8f c = {};
#pragma unroll
  for (int k0 = 0; k0 < EE; k0 += 32) {
    v16bf a = load_frag_rowptr(arow, k0, lane);
    v16bf b = load_frag(wihb, EE, nt * 16, k0, lane);
    c = wmma_bf16(a, b, c);
  }
  int col = nt * 16 + (lane & 15);
  float bias = bih[col];
  int row0 = mt * 16 + ((lane & 16) ? 8 : 0);
#pragma unroll
  for (int q = 0; q < 8; ++q)
    gx[(long long)(row0 + q) * 768 + col] = c[q] + bias;
}

// ---------------------------------------------------------------------------
// GRU: one workgroup per batch (batches independent -> no cross-WG sync).
// 768 threads; thread j owns w_hh row j (streamed bf16 from L2), h in LDS (f32).
// ---------------------------------------------------------------------------
__global__ void __launch_bounds__(768) gru_kernel(const float* __restrict__ gx,
                                                  const float* __restrict__ h0,
                                                  const unsigned short* __restrict__ whhb,
                                                  const float* __restrict__ bhh,
                                                  unsigned short* __restrict__ cat,
                                                  float* __restrict__ hT) {
  __shared__ float hbuf[HH];
  __shared__ float dots[3 * HH];
  int b = blockIdx.x;
  int j = threadIdx.x;

  float bh0 = 0.f, bh1 = 0.f, bh2 = 0.f;
  if (j < HH) {
    hbuf[j] = h0[b * HH + j];
    bh0 = bhh[j]; bh1 = bhh[HH + j]; bh2 = bhh[2 * HH + j];
  }
  __syncthreads();

  const uint4* wrow = (const uint4*)(whhb + (long long)j * HH);  // 32 x 16B

  for (int t = 0; t < TD; ++t) {
    float acc = 0.f;
#pragma unroll 8
    for (int q = 0; q < 32; ++q) {
      uint4 w = wrow[q];
      int k = q * 8;
      acc += bfbits2f((unsigned short)(w.x & 0xffff)) * hbuf[k + 0];
      acc += bfbits2f((unsigned short)(w.x >> 16))    * hbuf[k + 1];
      acc += bfbits2f((unsigned short)(w.y & 0xffff)) * hbuf[k + 2];
      acc += bfbits2f((unsigned short)(w.y >> 16))    * hbuf[k + 3];
      acc += bfbits2f((unsigned short)(w.z & 0xffff)) * hbuf[k + 4];
      acc += bfbits2f((unsigned short)(w.z >> 16))    * hbuf[k + 5];
      acc += bfbits2f((unsigned short)(w.w & 0xffff)) * hbuf[k + 6];
      acc += bfbits2f((unsigned short)(w.w >> 16))    * hbuf[k + 7];
    }
    dots[j] = acc;
    __syncthreads();
    if (j < HH) {
      const float* g = gx + (long long)(b * TD + t) * 768;
      float r = 1.f / (1.f + __expf(-(g[j] + dots[j] + bh0)));
      float z = 1.f / (1.f + __expf(-(g[HH + j] + dots[HH + j] + bh1)));
      float n = tanhf(g[2 * HH + j] + r * (dots[2 * HH + j] + bh2));
      float hn = (1.f - z) * n + z * hbuf[j];
      hbuf[j] = hn;
      cat[(long long)(b * TD + t) * 512 + j] = f2bf_bits(hn);
      if (t == TD - 1) hT[b * HH + j] = hn;
    }
    __syncthreads();
  }
}

// ---------------------------------------------------------------------------
// Attention: per (batch, 64-dec-column tile) workgroup.
//   S = hs_b @ out_b^T (WMMA), softmax over e, attn -> d_out,
//   c = attn^T @ hs_b (WMMA, A rebuilt from LDS columns) -> cat[:,256:512]
// ---------------------------------------------------------------------------
#define SS 68   // padded f32 LDS stride for the 256 x 64 score slab

__device__ __forceinline__ v16bf frag_from_S_cols(const float* S, int trow0, int k0,
                                                  int lane) {
  int t  = trow0 + (lane & 15);
  int ko = (lane & 16) ? 8 : 0;
  v16bf v;
#pragma unroll
  for (int j = 0; j < 8; ++j) {
    v[j]     = (__bf16)S[(k0 + ko + j) * SS + t];
    v[8 + j] = (__bf16)S[(k0 + 16 + ko + j) * SS + t];
  }
  return v;
}

__global__ void __launch_bounds__(256) attn_kernel(const unsigned short* __restrict__ hsb,
                                                   const unsigned short* __restrict__ hsTb,
                                                   const unsigned short* __restrict__ catr,
                                                   float* __restrict__ attn_out,
                                                   unsigned short* __restrict__ catw) {
  __shared__ float S[TE * SS];   // 256 x 64 scores (padded)
  int b    = blockIdx.y;
  int t0   = blockIdx.x * 64;
  int lane = threadIdx.x & 31;
  int wave = threadIdx.x >> 5;

  const unsigned short* hsrow = hsb + ((long long)b << 16);       // (e,h) stride 256
  const unsigned short* outb  = catr + (long long)b * TD * 512;   // rows t, stride 512

  // ---- scores: s[e,t] = sum_h hs[e,h]*out[t,h]
#pragma unroll
  for (int ei = 0; ei < 2; ++ei) {
    int et = wave * 2 + ei;                // 16 e-tiles over 8 waves
#pragma unroll
    for (int tt = 0; tt < 4; ++tt) {
      v8f c = {};
#pragma unroll
      for (int k0 = 0; k0 < HH; k0 += 32) {
        v16bf a = load_frag(hsrow, HH, et * 16, k0, lane);
        v16bf bb = load_frag(outb, 512, t0 + tt * 16, k0, lane);
        c = wmma_bf16(a, bb, c);
      }
      int tc = tt * 16 + (lane & 15);
      int hi = (lane & 16) ? 8 : 0;
#pragma unroll
      for (int q = 0; q < 8; ++q) S[(et * 16 + q + hi) * SS + tc] = c[q];
    }
  }
  __syncthreads();

  // ---- softmax over e (one thread per dec column; bank-conflict-free: pad 68)
  if (threadIdx.x < 64) {
    int t = threadIdx.x;
    float m = -1e30f;
    for (int e = 0; e < TE; ++e) m = fmaxf(m, S[e * SS + t]);
    float s = 0.f;
    for (int e = 0; e < TE; ++e) { float v = __expf(S[e * SS + t] - m); S[e * SS + t] = v; s += v; }
    float inv = 1.f / s;
    for (int e = 0; e < TE; ++e) S[e * SS + t] *= inv;
  }
  __syncthreads();

  // ---- write attn (b, e, t) f32, coalesced
  for (int i = threadIdx.x; i < TE * 64; i += 256) {
    int e = i >> 6, c = i & 63;
    attn_out[((long long)b * TE + e) * TD + t0 + c] = S[e * SS + c];
  }

  // ---- c[t,h] = sum_e attn[e,t] * hs[e,h]   (A = attn^T from LDS, B = hsT)
  int mt  = wave >> 1;                 // 4 t-tiles
  int nlo = (wave & 1) * 8;            // 16 h-tiles split 2 ways
  const unsigned short* hsT = hsTb + ((long long)b << 16);   // (h,e) stride 256

  v16bf afr[8];
#pragma unroll
  for (int kk = 0; kk < 8; ++kk) afr[kk] = frag_from_S_cols(S, mt * 16, kk * 32, lane);

#pragma unroll
  for (int nn = 0; nn < 8; ++nn) {
    int ntile = nlo + nn;
    v8f c = {};
#pragma unroll
    for (int kk = 0; kk < 8; ++kk) {
      v16bf bb = load_frag(hsT, TE, ntile * 16, kk * 32, lane);
      c = wmma_bf16(afr[kk], bb, c);
    }
    int h  = ntile * 16 + (lane & 15);
    int hi = (lane & 16) ? 8 : 0;
#pragma unroll
    for (int q = 0; q < 8; ++q)
      catw[(long long)(b * TD + t0 + mt * 16 + q + hi) * 512 + 256 + h] = f2bf_bits(c[q]);
  }
}

// ---------------------------------------------------------------------------
// logits = cat(8192x512) @ w_out^T(32000x512) + b_out
// 128x128 tile per WG, K=512 fully LDS-resident via TDM, 512 threads (16 waves),
// each wave 2x2 C-tiles. LDS row stride 520 bf16 (TDM pad) -> conflict-free b128.
// ---------------------------------------------------------------------------
#define LSTR 520

__global__ void __launch_bounds__(512) logits_kernel(const unsigned short* __restrict__ catb,
                                                     const unsigned short* __restrict__ woutb,
                                                     const float* __restrict__ bout,
                                                     float* __restrict__ out) {
  __shared__ __align__(16) unsigned short Ash[128 * LSTR];
  __shared__ __align__(16) unsigned short Bsh[128 * LSTR];

  long long m0 = (long long)blockIdx.y * 128;   // rows of cat
  long long n0 = (long long)blockIdx.x * 128;   // rows of w_out (logit cols)

  if ((threadIdx.x >> 5) == 0) {   // wave 0 issues both DMAs (EXEC-independent)
    tdm_load_tile_1024B_rows((unsigned)(uintptr_t)(void*)Ash, catb + m0 * 512, 128);
    tdm_load_tile_1024B_rows((unsigned)(uintptr_t)(void*)Bsh, woutb + n0 * 512, 128);
    __builtin_amdgcn_s_wait_tensorcnt(0);
  }
  __syncthreads();

  int lane = threadIdx.x & 31;
  int wave = threadIdx.x >> 5;
  int wm = wave >> 2, wn = wave & 3;           // 4x4 wave grid, 2x2 tiles each

  v8f c00 = {}, c01 = {}, c10 = {}, c11 = {};
#pragma unroll 4
  for (int k0 = 0; k0 < 512; k0 += 32) {
    v16bf a0 = load_frag(Ash, LSTR, (wm * 2) * 16, k0, lane);
    v16bf a1 = load_frag(Ash, LSTR, (wm * 2 + 1) * 16, k0, lane);
    v16bf b0 = load_frag(Bsh, LSTR, (wn * 2) * 16, k0, lane);
    v16bf b1 = load_frag(Bsh, LSTR, (wn * 2 + 1) * 16, k0, lane);
    c00 = wmma_bf16(a0, b0, c00);
    c01 = wmma_bf16(a0, b1, c01);
    c10 = wmma_bf16(a1, b0, c10);
    c11 = wmma_bf16(a1, b1, c11);
  }

  int cl = lane & 15;
  int hi = (lane & 16) ? 8 : 0;
  long long colA = n0 + (wn * 2) * 16 + cl;
  long long colB = n0 + (wn * 2 + 1) * 16 + cl;
  float biasA = bout[colA], biasB = bout[colB];
  long long rA = m0 + (wm * 2) * 16 + hi;
  long long rB = m0 + (wm * 2 + 1) * 16 + hi;
#pragma unroll
  for (int q = 0; q < 8; ++q) {
    out[(rA + q) * VV + colA] = c00[q] + biasA;
    out[(rA + q) * VV + colB] = c01[q] + biasB;
    out[(rB + q) * VV + colA] = c10[q] + biasA;
    out[(rB + q) * VV + colB] = c11[q] + biasB;
  }
}

// ---------------------------------------------------------------------------
extern "C" void kernel_launch(void* const* d_in, const int* in_sizes, int n_in,
                              void* d_out, int out_size, void* d_ws, size_t ws_size,
                              hipStream_t stream) {
  const int*   seq  = (const int*)d_in[0];
  const float* hs   = (const float*)d_in[1];
  const float* h0   = (const float*)d_in[2];
  const float* emb  = (const float*)d_in[3];
  const float* wih  = (const float*)d_in[4];
  const float* whh  = (const float*)d_in[5];
  const float* bih  = (const float*)d_in[6];
  const float* bhh  = (const float*)d_in[7];
  const float* wout = (const float*)d_in[8];
  const float* bout = (const float*)d_in[9];

  char* ws = (char*)d_ws;
  float*          gxbuf = (float*)ws;                               // 25,165,824 B
  unsigned short* cat   = (unsigned short*)(ws + 25165824);         //  8,388,608 B
  unsigned short* embb  = (unsigned short*)(ws + 33554432);         //  8,192,000 B
  unsigned short* wihb  = (unsigned short*)(ws + 41746432);         //    196,608 B
  unsigned short* whhb  = (unsigned short*)(ws + 41943040);         //    393,216 B
  unsigned short* woutb = (unsigned short*)(ws + 42336256);         // 32,768,000 B
  unsigned short* hsb   = (unsigned short*)(ws + 75104256);         //  4,194,304 B
  unsigned short* hsTb  = (unsigned short*)(ws + 79298560);         //  4,194,304 B

  float* out_logits = (float*)d_out;                  // 262,144,000 f32
  float* out_hT     = out_logits + 262144000ll;       //       8,192 f32
  float* out_attn   = out_hT + 8192;                  //   2,097,152 f32

  // precision staging: bf16 copies for WMMA operands
  convert_bf16<<<(VV * EE + 255) / 256, 256, 0, stream>>>(emb, embb, VV * EE);
  convert_bf16<<<(768 * EE + 255) / 256, 256, 0, stream>>>(wih, wihb, 768 * EE);
  convert_bf16<<<(768 * HH + 255) / 256, 256, 0, stream>>>(whh, whhb, 768 * HH);
  convert_bf16<<<(VV * 512 + 255) / 256, 256, 0, stream>>>(wout, woutb, VV * 512);
  convert_bf16<<<(NB * TE * HH + 255) / 256, 256, 0, stream>>>(hs, hsb, NB * TE * HH);
  transpose_hs_kernel<<<(NB * TE * HH + 255) / 256, 256, 0, stream>>>(hs, hsTb);

  gx_kernel<<<dim3(48, 64), 256, 0, stream>>>(seq, embb, wihb, bih, gxbuf);
  gru_kernel<<<NB, 768, 0, stream>>>(gxbuf, h0, whhb, bhh, cat, out_hT);
  attn_kernel<<<dim3(4, NB), 256, 0, stream>>>(hsb, hsTb, cat, out_attn, cat);
  logits_kernel<<<dim3(VV / 128, 8192 / 128), 512, 0, stream>>>(cat, woutb, bout,
                                                                out_logits);
}
